// WindowAttention_12807592477288
// MI455X (gfx1250) — matrix operand order repaired and checked by reading, not compiled, not run on hardware
//
#include <hip/hip_runtime.h>

typedef _Float16 half_t;
typedef __attribute__((ext_vector_type(16))) _Float16 v16h;
typedef __attribute__((ext_vector_type(8)))  _Float16 v8h;
typedef __attribute__((ext_vector_type(8)))  float    v8f;
typedef __attribute__((ext_vector_type(4)))  float    v4f;

#define DIMC   512
#define HEADS  16
#define HD     32
#define NTOK   64
#define NWIN   1568
#define QKVF   (3 * DIMC)          // 1536

// LDS layout (units: halfs). Rows padded to keep 16B alignment and rotate banks.
#define XS        (DIMC + 8)                  // 520 halfs row stride (x/q/k/o)
#define VTS       (NTOK + 8)                  // 72 halfs row stride (v^T)
#define PS        (NTOK + 8)                  // 72 halfs row stride (P)
#define OFF_XO    0                           // x, later reused for O
#define OFF_Q     (OFF_XO + NTOK * XS)        // 33280
#define OFF_K     (OFF_Q  + NTOK * XS)        // 66560
#define OFF_VT    (OFF_K  + NTOK * XS)        // 99840
#define OFF_P     (OFF_VT + DIMC * VTS)       // 136704
#define LDS_HALFS (OFF_P + 16 * 16 * PS)      // 155136 halfs = 310272 B (< 320KB)

__device__ __forceinline__ v8f wmma16(v16h a, v16h b, v8f c) {
  return __builtin_amdgcn_wmma_f32_16x16x32_f16(false, a, false, b, (short)0, c,
                                                false, false);
}

// A-matrix fragment (16x32 f16) from a row-major buffer.
// ISA 7.12.2: lane = half*16 + m; VGPR0-3 hold K {0..7 | 8..15}, VGPR4-7 hold
// K {16..23 | 24..31}.
__device__ __forceinline__ v16h frag_a(const half_t* base, int stride, int m0,
                                       int k0, int lane) {
  const int m  = m0 + (lane & 15);
  const int hs = lane >> 4;
  const half_t* p = base + m * stride + k0 + hs * 8;
  v8h lo = *(const v8h*)p;
  v8h hi = *(const v8h*)(p + 16);
  return __builtin_shufflevector(lo, hi, 0, 1, 2, 3, 4, 5, 6, 7,
                                 8, 9, 10, 11, 12, 13, 14, 15);
}

// B-matrix fragment (32x16 f16) where B[k][n] = W[n][k], W row-major.
// Same K order as frag_a: lane = half*16 + n; elements 0-7 hold K {0..7 | 8..15},
// elements 8-15 hold K {16..23 | 24..31}.
__device__ __forceinline__ v16h frag_b(const half_t* base, int stride, int n0,
                                       int k0, int lane) {
  const int n  = n0 + (lane & 15);
  const int hs = lane >> 4;
  const half_t* p = base + n * stride + k0 + hs * 8;
  v8h lo = *(const v8h*)p;
  v8h hi = *(const v8h*)(p + 16);
  return __builtin_shufflevector(lo, hi, 0, 1, 2, 3, 4, 5, 6, 7,
                                 8, 9, 10, 11, 12, 13, 14, 15);
}

// ---- kernel 1: convert weights to f16 (runs once per launch, tiny) ----
__global__ void wa_convert_w(const float* __restrict__ qkvw,
                             const float* __restrict__ projw,
                             half_t* __restrict__ qh,
                             half_t* __restrict__ ph) {
  const int i = blockIdx.x * blockDim.x + threadIdx.x;
  const int nq = QKVF * DIMC;        // 786432
  if (i < nq) {
    qh[i] = (half_t)qkvw[i];
  } else {
    const int j = i - nq;            // < 262144
    ph[j] = (half_t)projw[j];
  }
}

// ---- kernel 2: pre-gather relative position bias -> bias[h][n][m] ----
__global__ void wa_build_bias(const float* __restrict__ table,
                              const int* __restrict__ rel_idx,
                              float* __restrict__ bias) {
  const int i  = blockIdx.x * blockDim.x + threadIdx.x;   // 65536 threads
  const int h  = i >> 12;
  const int nm = i & 4095;
  bias[i] = table[rel_idx[nm] * HEADS + h];
}

// ---- kernel 3: fully fused window attention, one workgroup per window ----
extern __shared__ half_t smem[];

__global__ __launch_bounds__(512, 1)
void wa_fused(const float* __restrict__ x,
              const half_t* __restrict__ qkvw,
              const half_t* __restrict__ projw,
              const float* __restrict__ bias,
              const float* __restrict__ proj_b,
              float* __restrict__ out) {
  const int win  = blockIdx.x;
  const int tid  = threadIdx.x;
  const int lane = tid & 31;
  const int wave = tid >> 5;          // 16 waves

  half_t* xo = smem + OFF_XO;         // x (phase1), then O (phase2/3)
  half_t* qb = smem + OFF_Q;
  half_t* kb = smem + OFF_K;
  half_t* vt = smem + OFF_VT;         // v transposed: [f=h*32+d][token]
  half_t* pw = smem + OFF_P + wave * (16 * PS);

  // ---- Phase 0: x fp32 -> LDS f16 ----
  {
    const float* xg = x + (size_t)win * NTOK * DIMC;
    const int row = tid >> 3;              // 64 rows, 8 threads per row
    const int cb  = (tid & 7) * 64;        // 64 cols per thread
#pragma unroll
    for (int c = 0; c < 64; c += 8) {
      v4f a = *(const v4f*)(xg + row * DIMC + cb + c);
      v4f b = *(const v4f*)(xg + row * DIMC + cb + c + 4);
      v8h hv;
      hv[0] = (half_t)a[0]; hv[1] = (half_t)a[1];
      hv[2] = (half_t)a[2]; hv[3] = (half_t)a[3];
      hv[4] = (half_t)b[0]; hv[5] = (half_t)b[1];
      hv[6] = (half_t)b[2]; hv[7] = (half_t)b[3];
      *(v8h*)(xo + row * XS + cb + c) = hv;
    }
  }
  __syncthreads();

  // ---- Phase 1: QKV = x @ qkv_w^T ; write q(scaled)/k row-major, v^T ----
  {
    const int rt = wave & 3;          // token row tile (16 rows)
    const int cg = wave >> 2;         // column group
    const int m0 = rt * 16;
    const int col = lane & 15;
    const int hs  = lane >> 4;
#pragma unroll 2
    for (int j = 0; j < 12; ++j) {    // 2 col-tiles per iter -> 24 tiles/wave
      const int ct0 = cg + 8 * j;
      const int ct1 = ct0 + 4;
      v8f acc0 = {}; v8f acc1 = {};
#pragma unroll 4
      for (int ks = 0; ks < 16; ++ks) {
        v16h af = frag_a(xo, XS, m0, ks * 32, lane);
        v16h b0 = frag_b(qkvw, DIMC, ct0 * 16, ks * 32, lane);
        v16h b1 = frag_b(qkvw, DIMC, ct1 * 16, ks * 32, lane);
        acc0 = wmma16(af, b0, acc0);
        acc1 = wmma16(af, b1, acc1);
      }
      // scatter the two tiles to q/k/v buffers
      const int f0a[2] = {ct0 * 16, ct1 * 16};
      const v8f* accs[2] = {&acc0, &acc1};
#pragma unroll
      for (int t = 0; t < 2; ++t) {
        const int f0 = f0a[t];
        const v8f av = *accs[t];
        if (f0 < DIMC) {                       // Q, fold in 1/sqrt(hd)
          const float s = 0.17677669529663689f;
          const int f = f0 + col;
#pragma unroll
          for (int v = 0; v < 8; ++v)
            qb[(m0 + v + 8 * hs) * XS + f] = (half_t)(av[v] * s);
        } else if (f0 < 2 * DIMC) {            // K
          const int f = f0 - DIMC + col;
#pragma unroll
          for (int v = 0; v < 8; ++v)
            kb[(m0 + v + 8 * hs) * XS + f] = (half_t)av[v];
        } else {                               // V, store transposed
          const int f = f0 - 2 * DIMC + col;   // = h*32 + d
          v8h hv;
#pragma unroll
          for (int v = 0; v < 8; ++v) hv[v] = (half_t)av[v];
          *(v8h*)(vt + f * VTS + m0 + 8 * hs) = hv;
        }
      }
    }
  }
  __syncthreads();

  // ---- Phase 2: per-head attention (wave w owns head w) ----
  {
    const int h   = wave;
    const int col = lane & 15;
    const int hs  = lane >> 4;
    const float* bh = bias + h * (NTOK * NTOK);
    for (int r = 0; r < 4; ++r) {
      const int n0 = r * 16;
      // S = (q/scale) @ k^T   [16 x 64] as 4 tiles
      v16h aq = frag_a(qb, XS, n0, h * HD, lane);
      v8f s[4];
#pragma unroll
      for (int ct = 0; ct < 4; ++ct) {
        v16h bk = frag_b(kb, XS, ct * 16, h * HD, lane);
        v8f z = {};
        s[ct] = wmma16(aq, bk, z);
      }
      // + relative position bias
#pragma unroll
      for (int ct = 0; ct < 4; ++ct)
#pragma unroll
        for (int v = 0; v < 8; ++v)
          s[ct][v] += bh[(n0 + v + 8 * hs) * NTOK + ct * 16 + col];
      // row-wise softmax: row (v, half) lives in 16 lanes x 4 regs
#pragma unroll
      for (int v = 0; v < 8; ++v) {
        float mx = fmaxf(fmaxf(s[0][v], s[1][v]), fmaxf(s[2][v], s[3][v]));
#pragma unroll
        for (int o = 8; o >= 1; o >>= 1) mx = fmaxf(mx, __shfl_xor(mx, o, 32));
        float sum = 0.f;
#pragma unroll
        for (int ct = 0; ct < 4; ++ct) {
          float e = __expf(s[ct][v] - mx);
          s[ct][v] = e; sum += e;
        }
#pragma unroll
        for (int o = 8; o >= 1; o >>= 1) sum += __shfl_xor(sum, o, 32);
        const float inv = 1.0f / sum;
#pragma unroll
        for (int ct = 0; ct < 4; ++ct) s[ct][v] *= inv;
      }
      // P (C-layout) -> per-wave LDS slab (A-layout source)
#pragma unroll
      for (int ct = 0; ct < 4; ++ct)
#pragma unroll
        for (int v = 0; v < 8; ++v)
          pw[(v + 8 * hs) * PS + ct * 16 + col] = (half_t)s[ct][v];
      // O = P @ V  (K = 64 in two chunks; two 16-wide d tiles)
      v8f o0 = {}, o1 = {};
#pragma unroll
      for (int mc = 0; mc < 2; ++mc) {
        v16h ap  = frag_a(pw, PS, 0, mc * 32, lane);
        v16h bv0 = frag_b(vt, VTS, h * HD, mc * 32, lane);
        v16h bv1 = frag_b(vt, VTS, h * HD + 16, mc * 32, lane);
        o0 = wmma16(ap, bv0, o0);
        o1 = wmma16(ap, bv1, o1);
      }
      // store O into the (former x) LDS region, feature = h*32 + d
#pragma unroll
      for (int v = 0; v < 8; ++v) {
        const int tok = n0 + v + 8 * hs;
        xo[tok * XS + h * HD + col]      = (half_t)o0[v];
        xo[tok * XS + h * HD + 16 + col] = (half_t)o1[v];
      }
    }
  }
  __syncthreads();

  // ---- Phase 3: out = O @ proj_w^T + proj_b (fp32 out) ----
  {
    const int rt = wave & 3;
    const int cg = wave >> 2;
    const int m0 = rt * 16;
    const int col = lane & 15;
    const int hs  = lane >> 4;
    float* og = out + (size_t)win * NTOK * DIMC;
#pragma unroll 2
    for (int j = 0; j < 4; ++j) {     // 2 col-tiles per iter -> 8 tiles/wave
      const int ct0 = cg + 8 * j;
      const int ct1 = ct0 + 4;
      v8f a0 = {}; v8f a1 = {};
#pragma unroll 4
      for (int ks = 0; ks < 16; ++ks) {
        v16h af = frag_a(xo, XS, m0, ks * 32, lane);
        v16h b0 = frag_b(projw, DIMC, ct0 * 16, ks * 32, lane);
        v16h b1 = frag_b(projw, DIMC, ct1 * 16, ks * 32, lane);
        a0 = wmma16(af, b0, a0);
        a1 = wmma16(af, b1, a1);
      }
      const float bb0 = proj_b[ct0 * 16 + col];
      const float bb1 = proj_b[ct1 * 16 + col];
#pragma unroll
      for (int v = 0; v < 8; ++v) {
        const int tok = m0 + v + 8 * hs;
        og[tok * DIMC + ct0 * 16 + col] = a0[v] + bb0;
        og[tok * DIMC + ct1 * 16 + col] = a1[v] + bb1;
      }
    }
  }
}

extern "C" void kernel_launch(void* const* d_in, const int* in_sizes, int n_in,
                              void* d_out, int out_size, void* d_ws, size_t ws_size,
                              hipStream_t stream) {
  (void)in_sizes; (void)n_in; (void)out_size; (void)ws_size;
  const float* x      = (const float*)d_in[0];
  const float* qkv_w  = (const float*)d_in[1];
  const float* table  = (const float*)d_in[2];
  const float* proj_w = (const float*)d_in[3];
  const float* proj_b = (const float*)d_in[4];
  const int*   rel_ix = (const int*)d_in[5];
  float* out = (float*)d_out;

  char* ws = (char*)d_ws;
  half_t* qkvw_h = (half_t*)ws;                                  // 1,572,864 B
  half_t* projw_h = (half_t*)(ws + (size_t)QKVF * DIMC * 2);     //   524,288 B
  float*  bias    = (float*)(ws + (size_t)QKVF * DIMC * 2
                                + (size_t)DIMC * DIMC * 2);      //   262,144 B

  // weights -> f16 (786432 + 262144 = 1,048,576 elems)
  wa_convert_w<<<4096, 256, 0, stream>>>(qkv_w, proj_w, qkvw_h, projw_h);
  // bias[h][n][m] gather (65,536 elems)
  wa_build_bias<<<256, 256, 0, stream>>>(table, rel_ix, bias);

  const size_t lds_bytes = (size_t)LDS_HALFS * sizeof(half_t);   // 310,272 B
  static_assert(LDS_HALFS * sizeof(half_t) <= 320u * 1024u, "LDS budget");
  (void)hipFuncSetAttribute((const void*)wa_fused,
                            hipFuncAttributeMaxDynamicSharedMemorySize,
                            (int)lds_bytes);
  wa_fused<<<NWIN, 512, lds_bytes, stream>>>(x, qkvw_h, projw_h, bias,
                                             proj_b, out);
}
